// SpatialMambaBlock_7086696038759
// MI455X (gfx1250) — compile-verified, matrix operand order checked by reading, not run on hardware
//
#include <hip/hip_runtime.h>
#include <hip/hip_bf16.h>

// ---------------------------------------------------------------------------
// Bimamba block for MI455X (gfx1250, wave32, WMMA).
//  * GEMMs: v_wmma_f32_16x16x32_bf16, double-buffered LDS, register-prefetch
//    software pipeline (fp32 -> bf16 conversion during LDS staging).
//  * Selective scan: chunked parallel scan (8 chunks x 64 steps); chunk B/C
//    staged to LDS via gfx1250 async global->LDS copy when available.
// ---------------------------------------------------------------------------

#define D_MODEL   1024
#define D_STATE   16
#define D_CONV    4
#define D_INNER   2048
#define DT_RANK   64
#define B_SZ      2
#define SEQ_L     512
#define TOKENS    (B_SZ * SEQ_L)   // 1024
#define NCHUNK    8
#define CHUNK     64               // SEQ_L / NCHUNK
#define XPROJ_N   (DT_RANK + 2 * D_STATE)   // 96

typedef __attribute__((ext_vector_type(16))) __bf16 bf16x16;
typedef __attribute__((ext_vector_type(8)))  __bf16 bf16x8;
typedef __attribute__((ext_vector_type(8)))  float  v8f;

#define LDS_STRIDE 40   // bf16 elems per LDS tile row: 80B, 16B-aligned chunks

// ---- optional gfx1250 async global->LDS copy (guarded; falls back) --------
#if defined(__AMDGCN__) && \
    __has_builtin(__builtin_amdgcn_global_load_async_to_lds_b128) && \
    __has_builtin(__builtin_amdgcn_s_wait_asynccnt)
#define USE_ASYNC_LDS 1
#endif

typedef int v4i_vs __attribute__((vector_size(4 * sizeof(int))));

__device__ __forceinline__ void stage16B(const float* __restrict__ g, float* lds)
{
#if defined(USE_ASYNC_LDS)
    __builtin_amdgcn_global_load_async_to_lds_b128(
        (__attribute__((address_space(1))) v4i_vs*)g,
        (__attribute__((address_space(3))) v4i_vs*)lds,
        0, 0);
#else
    *(float4*)lds = *(const float4*)g;
#endif
}

__device__ __forceinline__ void stage_wait()
{
#if defined(USE_ASYNC_LDS)
    __builtin_amdgcn_s_wait_asynccnt(0);
#endif
}

// ---------------------------------------------------------------------------
// Tiled WMMA GEMM:  C[M,N] = A[M,K] * W[N,K]^T   (fp32 in memory, bf16 math)
// 256 threads / 8 waves; block tile 128x64; wave tile 32x32 (2x2 WMMA frags);
// K step 32; double-buffered LDS with register prefetch.
// epilogue: 0 = none, 1 = softplus(x + bias[n])
// ---------------------------------------------------------------------------
__global__ __launch_bounds__(256)
void wmma_gemm_bf16(const float* __restrict__ A, int lda,
                    const float* __restrict__ W, int ldb,
                    float* __restrict__ C, int ldc,
                    int M, int N, int K,
                    const float* __restrict__ bias, int epilogue)
{
    __shared__ __align__(16) __bf16 As[2][128 * LDS_STRIDE];
    __shared__ __align__(16) __bf16 Bs[2][64 * LDS_STRIDE];

    const int tid  = threadIdx.x;
    const int lane = tid & 31;
    const int wave = tid >> 5;
    const int wm   = wave >> 1;       // 0..3
    const int wn   = wave & 1;        // 0..1
    const int half = lane >> 4;
    const int l16  = lane & 15;

    const int mBlock = blockIdx.y * 128;
    const int nBlock = blockIdx.x * 64;

    float4 ra[4], rb[2];

    auto ldglobal = [&](int k0) {
        #pragma unroll
        for (int s = 0; s < 4; ++s) {
            int slot = tid + s * 256;
            int r    = slot >> 3;
            int c4   = (slot & 7) << 2;
            ra[s] = *(const float4*)(A + (size_t)(mBlock + r) * lda + k0 + c4);
        }
        #pragma unroll
        for (int s = 0; s < 2; ++s) {
            int slot = tid + s * 256;
            int r    = slot >> 3;
            int c4   = (slot & 7) << 2;
            int n    = nBlock + r;
            rb[s] = (n < N) ? *(const float4*)(W + (size_t)n * ldb + k0 + c4)
                            : make_float4(0.f, 0.f, 0.f, 0.f);
        }
    };

    auto stash = [&](int buf) {
        #pragma unroll
        for (int s = 0; s < 4; ++s) {
            int slot = tid + s * 256;
            int r    = slot >> 3;
            int c4   = (slot & 7) << 2;
            __bf16* dst = &As[buf][r * LDS_STRIDE + c4];
            dst[0] = (__bf16)ra[s].x; dst[1] = (__bf16)ra[s].y;
            dst[2] = (__bf16)ra[s].z; dst[3] = (__bf16)ra[s].w;
        }
        #pragma unroll
        for (int s = 0; s < 2; ++s) {
            int slot = tid + s * 256;
            int r    = slot >> 3;
            int c4   = (slot & 7) << 2;
            __bf16* dst = &Bs[buf][r * LDS_STRIDE + c4];
            dst[0] = (__bf16)rb[s].x; dst[1] = (__bf16)rb[s].y;
            dst[2] = (__bf16)rb[s].z; dst[3] = (__bf16)rb[s].w;
        }
    };

    v8f acc[2][2] = {};
    const int nk = K >> 5;

    ldglobal(0);
    for (int kk = 0; kk < nk; ++kk) {
        const int buf = kk & 1;
        stash(buf);
        __syncthreads();
        if (kk + 1 < nk) ldglobal((kk + 1) << 5);   // overlap with WMMAs below
        if (kk + 2 < nk)                            // L2 prefetch two tiles out
            __builtin_prefetch(A + (size_t)(mBlock + (tid >> 1)) * lda + ((kk + 2) << 5), 0, 0);

        // fragments: lane<16 -> K[0..7],K[16..23]; lane>=16 -> K[8..15],K[24..31]
        bf16x16 afrag[2], bfrag[2];
        #pragma unroll
        for (int i = 0; i < 2; ++i) {
            int row = (wm * 32 + i * 16 + l16) * LDS_STRIDE;
            bf16x8 lo = *(const bf16x8*)(&As[buf][row + 8 * half]);
            bf16x8 hi = *(const bf16x8*)(&As[buf][row + 16 + 8 * half]);
            bf16x16 a;
            #pragma unroll
            for (int e = 0; e < 8; ++e) { a[e] = lo[e]; a[e + 8] = hi[e]; }
            afrag[i] = a;
        }
        #pragma unroll
        for (int j = 0; j < 2; ++j) {
            int row = (wn * 32 + j * 16 + l16) * LDS_STRIDE;
            bf16x8 lo = *(const bf16x8*)(&Bs[buf][row + 8 * half]);
            bf16x8 hi = *(const bf16x8*)(&Bs[buf][row + 16 + 8 * half]);
            bf16x16 b;
            #pragma unroll
            for (int e = 0; e < 8; ++e) { b[e] = lo[e]; b[e + 8] = hi[e]; }
            bfrag[j] = b;
        }

        #pragma unroll
        for (int i = 0; i < 2; ++i)
            #pragma unroll
            for (int j = 0; j < 2; ++j)
                acc[i][j] = __builtin_amdgcn_wmma_f32_16x16x32_bf16(
                    false, afrag[i], false, bfrag[j],
                    (short)0, acc[i][j], false, false);
        __syncthreads();
    }

    // epilogue + store: VGPR r -> M = r + 8*half, N = lane&15
    #pragma unroll
    for (int i = 0; i < 2; ++i) {
        #pragma unroll
        for (int j = 0; j < 2; ++j) {
            int mBase = mBlock + wm * 32 + i * 16 + half * 8;
            int nCol  = nBlock + wn * 32 + j * 16 + l16;
            if (nCol >= N) continue;
            float bval = (epilogue == 1) ? bias[nCol] : 0.f;
            #pragma unroll
            for (int r = 0; r < 8; ++r) {
                float v = acc[i][j][r];
                if (epilogue == 1) {
                    v += bval;
                    v = (v > 20.f) ? v : log1pf(__expf(v));   // softplus
                }
                C[(size_t)(mBase + r) * ldc + nCol] = v;
            }
        }
    }
}

// ---------------------------------------------------------------------------
// Depthwise causal conv (k=4) + SiLU.
// ---------------------------------------------------------------------------
__global__ __launch_bounds__(256)
void dwconv_silu(const float* __restrict__ x, int x_ld,
                 const float* __restrict__ w, const float* __restrict__ b,
                 float* __restrict__ y, int total, int L, int D)
{
    int idx = blockIdx.x * 256 + threadIdx.x;
    if (idx >= total) return;
    int d  = idx % D;
    int l  = (idx / D) % L;
    int bi = idx / (D * L);
    float acc = b[d];
    #pragma unroll
    for (int j = 0; j < D_CONV; ++j) {
        int ls = l - (D_CONV - 1) + j;
        if (ls >= 0)
            acc += w[d * D_CONV + j] * x[(size_t)(bi * L + ls) * x_ld + d];
    }
    y[idx] = acc * (1.f / (1.f + __expf(-acc)));
}

// ---------------------------------------------------------------------------
// Chunked selective scan.
// ---------------------------------------------------------------------------
__device__ __forceinline__ int tok_of(int b, int cl, int s, int reverse)
{
    int sl = cl * CHUNK + s;                 // position in scan order
    int l  = reverse ? (SEQ_L - 1 - sl) : sl;
    return b * SEQ_L + l;
}

// Pass 1: per-chunk local end state H and chunk decay P = exp(A * sum(dt)).
__global__ __launch_bounds__(256)
void scan_chunk_state(const float* __restrict__ u,
                      const float* __restrict__ delta,
                      const float* __restrict__ xdbl,
                      const float* __restrict__ A_log,
                      float* __restrict__ H, float* __restrict__ P,
                      int reverse)
{
    __shared__ __align__(16) float sB[CHUNK * D_STATE];   // 4 KB
    const int tid = threadIdx.x;
    const int d   = blockIdx.x * 256 + tid;
    const int cl  = blockIdx.y;
    const int b   = blockIdx.z;

    {   // 1024 floats = 256 float4 slots, one per thread
        int s  = tid >> 2;
        int j4 = (tid & 3) << 2;
        stage16B(&xdbl[(size_t)tok_of(b, cl, s, reverse) * XPROJ_N + DT_RANK + j4],
                 &sB[s * D_STATE + j4]);
    }
    stage_wait();
    __syncthreads();

    float Arow[D_STATE];
    #pragma unroll
    for (int n = 0; n < D_STATE; ++n)
        Arow[n] = -__expf(A_log[(size_t)d * D_STATE + n]);

    float h[D_STATE];
    #pragma unroll
    for (int n = 0; n < D_STATE; ++n) h[n] = 0.f;
    float dtsum = 0.f;

    for (int s = 0; s < CHUNK; ++s) {
        const int t = tok_of(b, cl, s, reverse);
        const float dt = delta[(size_t)t * D_INNER + d];
        const float du = dt * u[(size_t)t * D_INNER + d];
        dtsum += dt;
        #pragma unroll
        for (int n = 0; n < D_STATE; ++n)
            h[n] = h[n] * __expf(dt * Arow[n]) + du * sB[s * D_STATE + n];
    }

    const size_t o = ((size_t)(b * NCHUNK + cl) * D_INNER + d) * D_STATE;
    #pragma unroll
    for (int n = 0; n < D_STATE; ++n) {
        H[o + n] = h[n];
        P[o + n] = __expf(Arow[n] * dtsum);
    }
}

// Pass 2: sequential combine over the 8 chunks -> incoming state per chunk.
__global__ __launch_bounds__(256)
void scan_chunk_combine(const float* __restrict__ H,
                        const float* __restrict__ P,
                        float* __restrict__ Gin)
{
    const int idx = blockIdx.x * 256 + threadIdx.x;   // over B * D_INNER * 16
    const int rem = idx & (D_INNER * D_STATE - 1);
    const int b   = idx >> 15;
    float g = 0.f;
    #pragma unroll
    for (int cl = 0; cl < NCHUNK; ++cl) {
        const size_t o = (size_t)(b * NCHUNK + cl) * (D_INNER * D_STATE) + rem;
        Gin[o] = g;
        g = g * P[o] + H[o];
    }
}

// Pass 3: rerun each chunk from its true incoming state; emit gated output.
__global__ __launch_bounds__(256)
void scan_chunk_output(const float* __restrict__ u,
                       const float* __restrict__ delta,
                       const float* __restrict__ xdbl,
                       const float* __restrict__ A_log,
                       const float* __restrict__ Dv,
                       const float* __restrict__ res, int res_ld,
                       const float* __restrict__ Gin,
                       float* __restrict__ out,
                       int reverse, int accum)
{
    __shared__ __align__(16) float sBC[CHUNK * 2 * D_STATE];   // 8 KB
    const int tid = threadIdx.x;
    const int d   = blockIdx.x * 256 + tid;
    const int cl  = blockIdx.y;
    const int b   = blockIdx.z;

    #pragma unroll
    for (int it = 0; it < 2; ++it) {       // 2048 floats = 512 float4 slots
        int idx = tid + it * 256;
        int s   = idx >> 3;
        int j4  = (idx & 7) << 2;
        stage16B(&xdbl[(size_t)tok_of(b, cl, s, reverse) * XPROJ_N + DT_RANK + j4],
                 &sBC[s * 2 * D_STATE + j4]);
    }
    stage_wait();
    __syncthreads();

    float Arow[D_STATE];
    #pragma unroll
    for (int n = 0; n < D_STATE; ++n)
        Arow[n] = -__expf(A_log[(size_t)d * D_STATE + n]);
    const float Dd = Dv[d];

    float h[D_STATE];
    const size_t o = ((size_t)(b * NCHUNK + cl) * D_INNER + d) * D_STATE;
    #pragma unroll
    for (int n = 0; n < D_STATE; ++n) h[n] = Gin[o + n];

    for (int s = 0; s < CHUNK; ++s) {
        const int t = tok_of(b, cl, s, reverse);
        const float dt = delta[(size_t)t * D_INNER + d];
        const float uu = u[(size_t)t * D_INNER + d];
        const float du = dt * uu;
        float y = 0.f;
        #pragma unroll
        for (int n = 0; n < D_STATE; ++n) {
            h[n] = h[n] * __expf(dt * Arow[n]) + du * sBC[s * 2 * D_STATE + n];
            y += h[n] * sBC[s * 2 * D_STATE + D_STATE + n];
        }
        const float rv   = res[(size_t)t * res_ld + d];
        const float gate = rv * (1.f / (1.f + __expf(-rv)));
        const float ov   = (y + uu * Dd) * gate;
        float* po = &out[(size_t)t * D_INNER + d];
        if (accum) *po += ov; else *po = ov;
    }
}

// ---------------------------------------------------------------------------
// Orchestration
// ---------------------------------------------------------------------------
extern "C" void kernel_launch(void* const* d_in, const int* in_sizes, int n_in,
                              void* d_out, int out_size, void* d_ws, size_t ws_size,
                              hipStream_t stream)
{
    const float* x           = (const float*)d_in[0];
    const float* in_proj_w   = (const float*)d_in[1];
    const float* conv_w      = (const float*)d_in[2];
    const float* conv_b      = (const float*)d_in[3];
    const float* x_proj_w    = (const float*)d_in[4];
    const float* dt_proj_w   = (const float*)d_in[5];
    const float* dt_proj_b   = (const float*)d_in[6];
    const float* A_log       = (const float*)d_in[7];
    const float* Dvec        = (const float*)d_in[8];
    const float* out_proj_w  = (const float*)d_in[9];
    const float* conv_b_w    = (const float*)d_in[10];
    const float* conv_b_b    = (const float*)d_in[11];
    const float* x_proj_b_w  = (const float*)d_in[12];
    const float* dt_proj_b_w = (const float*)d_in[13];
    const float* dt_proj_b_b = (const float*)d_in[14];
    const float* A_b_log     = (const float*)d_in[15];
    const float* D_b         = (const float*)d_in[16];

    float* ws      = (float*)d_ws;
    float* xr      = ws;                                        // [1024,4096]
    float* convf   = xr      + (size_t)TOKENS * 2 * D_INNER;    // [1024,2048]
    float* convb   = convf   + (size_t)TOKENS * D_INNER;
    float* xdbl_f  = convb   + (size_t)TOKENS * D_INNER;        // [1024,96]
    float* xdbl_b  = xdbl_f  + (size_t)TOKENS * XPROJ_N;
    float* delta_f = xdbl_b  + (size_t)TOKENS * XPROJ_N;        // [1024,2048]
    float* delta_b = delta_f + (size_t)TOKENS * D_INNER;
    float* ytot    = delta_b + (size_t)TOKENS * D_INNER;        // [1024,2048]
    float* Hbuf    = ytot    + (size_t)TOKENS * D_INNER;        // [2,8,2048,16]
    float* Pbuf    = Hbuf    + (size_t)B_SZ * NCHUNK * D_INNER * D_STATE;
    float* Gin     = Pbuf    + (size_t)B_SZ * NCHUNK * D_INNER * D_STATE;
    float* outp    = (float*)d_out;

    const dim3 blk(256);
    const dim3 scanGrid(D_INNER / 256, NCHUNK, B_SZ);

    // 1) in_proj
    wmma_gemm_bf16<<<dim3(2 * D_INNER / 64, TOKENS / 128), blk, 0, stream>>>(
        x, D_MODEL, in_proj_w, D_MODEL, xr, 2 * D_INNER,
        TOKENS, 2 * D_INNER, D_MODEL, nullptr, 0);

    // 2) fwd conv+silu on xf = xr[:, :2048]; 3) bwd conv+silu on convf
    dwconv_silu<<<dim3(TOKENS * D_INNER / 256), blk, 0, stream>>>(
        xr, 2 * D_INNER, conv_w, conv_b, convf, TOKENS * D_INNER, SEQ_L, D_INNER);
    dwconv_silu<<<dim3(TOKENS * D_INNER / 256), blk, 0, stream>>>(
        convf, D_INNER, conv_b_w, conv_b_b, convb, TOKENS * D_INNER, SEQ_L, D_INNER);

    // 4) x_proj fwd/bwd
    wmma_gemm_bf16<<<dim3((XPROJ_N + 63) / 64, TOKENS / 128), blk, 0, stream>>>(
        convf, D_INNER, x_proj_w, D_INNER, xdbl_f, XPROJ_N,
        TOKENS, XPROJ_N, D_INNER, nullptr, 0);
    wmma_gemm_bf16<<<dim3((XPROJ_N + 63) / 64, TOKENS / 128), blk, 0, stream>>>(
        convb, D_INNER, x_proj_b_w, D_INNER, xdbl_b, XPROJ_N,
        TOKENS, XPROJ_N, D_INNER, nullptr, 0);

    // 5) dt_proj fwd/bwd (fused bias+softplus)
    wmma_gemm_bf16<<<dim3(D_INNER / 64, TOKENS / 128), blk, 0, stream>>>(
        xdbl_f, XPROJ_N, dt_proj_w, DT_RANK, delta_f, D_INNER,
        TOKENS, D_INNER, DT_RANK, dt_proj_b, 1);
    wmma_gemm_bf16<<<dim3(D_INNER / 64, TOKENS / 128), blk, 0, stream>>>(
        xdbl_b, XPROJ_N, dt_proj_b_w, DT_RANK, delta_b, D_INNER,
        TOKENS, D_INNER, DT_RANK, dt_proj_b_b, 1);

    // 6) forward scan (chunked): write ytot
    scan_chunk_state<<<scanGrid, blk, 0, stream>>>(
        convf, delta_f, xdbl_f, A_log, Hbuf, Pbuf, 0);
    scan_chunk_combine<<<dim3(B_SZ * D_INNER * D_STATE / 256), blk, 0, stream>>>(
        Hbuf, Pbuf, Gin);
    scan_chunk_output<<<scanGrid, blk, 0, stream>>>(
        convf, delta_f, xdbl_f, A_log, Dvec, xr + D_INNER, 2 * D_INNER,
        Gin, ytot, 0, 0);

    // 7) backward scan (chunked, reversed): accumulate into ytot
    scan_chunk_state<<<scanGrid, blk, 0, stream>>>(
        convb, delta_b, xdbl_b, A_b_log, Hbuf, Pbuf, 1);
    scan_chunk_combine<<<dim3(B_SZ * D_INNER * D_STATE / 256), blk, 0, stream>>>(
        Hbuf, Pbuf, Gin);
    scan_chunk_output<<<scanGrid, blk, 0, stream>>>(
        convb, delta_b, xdbl_b, A_b_log, D_b, xr + D_INNER, 2 * D_INNER,
        Gin, ytot, 1, 1);

    // 8) out_proj -> d_out
    wmma_gemm_bf16<<<dim3(D_MODEL / 64, TOKENS / 128), blk, 0, stream>>>(
        ytot, D_INNER, out_proj_w, D_INNER, outp, D_MODEL,
        TOKENS, D_MODEL, D_INNER, nullptr, 0);
}
